// DemonstrationAttentionDotProductQModel_64175401337269
// MI455X (gfx1250) — matrix operand order repaired and checked by reading, not compiled, not run on hardware
//
#include <hip/hip_runtime.h>
#include <math.h>

// ---------------------------------------------------------------------------
// Model dims (fixed by the reference)
// ---------------------------------------------------------------------------
#define TT 64
#define BB 64
#define LL 100
#define DD 128
#define AA 18
#define HH 512
#define G4H 2048   // 4*H

typedef _Float16 half_t;
typedef __attribute__((ext_vector_type(16))) _Float16 v16h;
typedef __attribute__((ext_vector_type(8)))  _Float16 v8h;
typedef __attribute__((ext_vector_type(8)))  float    v8f;

#define DEV static __device__ __forceinline__

// ---------------------------------------------------------------------------
// WMMA fragment loaders (wave32, v_wmma_f32_16x16x32_f16)
//
// A (16x32, f16) per-lane layout (ISA 7.12.2):
//   lane l: row = l&15 ; holds K = {kb..kb+7} U {kb+16..kb+23}, kb = 8*(l>>4)
//   -> two contiguous 8-half (b128) loads from a row-major [M x K] matrix.
// B (32x16, f16): lane l: col = l&15 ; holds K = {kb..kb+7} U {kb+16..kb+23},
//   kb = 8*(l>>4) (same as A) -> two b128 loads from a row-major [N x K] matrix,
//   i.e. PyTorch [out,in] weights are already fragment-ready.
// ---------------------------------------------------------------------------
DEV v16h load_frag_a(const half_t* __restrict__ base, int ld, int row0, int k0, int lane) {
  const int r  = row0 + (lane & 15);
  const int kb = k0 + ((lane >> 4) << 3);
  const half_t* p = base + (size_t)r * ld + kb;
  v8h lo = *(const v8h*)p;
  v8h hi = *(const v8h*)(p + 16);
  v16h f;
#pragma unroll
  for (int i = 0; i < 8; ++i) { f[i] = lo[i]; f[i + 8] = hi[i]; }
  return f;
}

DEV v16h load_frag_b(const half_t* __restrict__ base, int ld, int col0, int k0, int lane) {
  const int n  = col0 + (lane & 15);
  const int kb = k0 + ((lane >> 4) << 3);
  const half_t* p = base + (size_t)n * ld + kb;
  v8h lo = *(const v8h*)p;
  v8h hi = *(const v8h*)(p + 16);
  v16h f;
#pragma unroll
  for (int i = 0; i < 8; ++i) { f[i] = lo[i]; f[i + 8] = hi[i]; }
  return f;
}

DEV v8f wmma_f16(v16h a, v16h b, v8f c) {
  // 8 args: (neg_a, A, neg_b, B, c_mod, C, reuse_a, reuse_b)
  return __builtin_amdgcn_wmma_f32_16x16x32_f16(false, a, false, b, (short)0, c,
                                                false, false);
}

DEV float sigmoidf_(float x) { return 1.0f / (1.0f + __expf(-x)); }

// ---------------------------------------------------------------------------
// Fused LSTM step: gates = A1*W1^T + A2*W2^T + bias ; then cell update.
// Grid: 32 WGs (one per 16 hidden columns), 512 threads = 16 waves.
// Wave w handles gate chunk g = w>>2, M-tile mt = w&3 (B=64 -> 4 M tiles).
// Gates staged in LDS; single barrier; elementwise cell update; h written
// as f32 (optional) and f16 (next step's WMMA operand — ping-pong buffer).
// ---------------------------------------------------------------------------
__global__ void lstm_step_kernel(
    const half_t* __restrict__ A1, int lda1, int k1,
    const half_t* __restrict__ W1,                 // [2048 x k1] row-major
    const half_t* __restrict__ A2,                 // [64 x 512] row-major
    const half_t* __restrict__ W2,                 // [2048 x 512]
    const float*  __restrict__ bias,               // [2048] (= bih+bhh)
    float*        __restrict__ c,                  // [64*512] in/out
    half_t*       __restrict__ h16out,             // [64*512]
    float*        __restrict__ h32out)             // optional
{
  __shared__ float gbuf[4][64][16];                // 16 KB
  const int lane = threadIdx.x & 31;
  const int wv   = threadIdx.x >> 5;
  const int jt   = blockIdx.x;                     // hidden-column tile 0..31
  const int g    = wv >> 2;
  const int mt   = wv & 3;
  const int n0   = g * HH + jt * 16;               // row in [2048 x K] weights

  v8f acc = {};
  for (int k0 = 0; k0 < k1; k0 += 32) {
    v16h a = load_frag_a(A1, lda1, mt * 16, k0, lane);
    v16h b = load_frag_b(W1, k1, n0, k0, lane);
    acc = wmma_f16(a, b, acc);
  }
  for (int k0 = 0; k0 < HH; k0 += 32) {
    v16h a = load_frag_a(A2, HH, mt * 16, k0, lane);
    v16h b = load_frag_b(W2, HH, n0, k0, lane);
    acc = wmma_f16(a, b, acc);
  }

  const float bv   = bias[n0 + (lane & 15)];
  const int   mrow = mt * 16 + ((lane >> 4) << 3);
#pragma unroll
  for (int r = 0; r < 8; ++r)
    gbuf[g][mrow + r][lane & 15] = acc[r] + bv;

  __syncthreads();

  for (int idx = threadIdx.x; idx < 64 * 16; idx += 512) {
    const int m = idx >> 4, jj = idx & 15;
    const int j = jt * 16 + jj;
    const float si = sigmoidf_(gbuf[0][m][jj]);
    const float sf = sigmoidf_(gbuf[1][m][jj]);
    const float tg = tanhf(gbuf[2][m][jj]);
    const float so = sigmoidf_(gbuf[3][m][jj]);
    const float cv = sf * c[m * HH + j] + si * tg;
    c[m * HH + j] = cv;
    const float hv = so * tanhf(cv);
    h16out[m * HH + j] = (half_t)hv;
    if (h32out) h32out[m * HH + j] = hv;
  }
}

// ---------------------------------------------------------------------------
// Generic WMMA GEMM: out[m,n] = sum_k A1[m,k]W1[n,k] (+ A2*W2^T) + bias[n]
// Grid-strided over 16x16 tiles, one tile per wave. Optional ReLU,
// f32 and/or f16 outputs, column clamp (nvalid) for the padded q head.
// ---------------------------------------------------------------------------
__global__ void gemm_wmma_kernel(
    const half_t* __restrict__ A1, int lda1, int k1,
    const half_t* __restrict__ W1, int ldw1,
    const half_t* __restrict__ A2, int lda2, int k2,
    const half_t* __restrict__ W2, int ldw2,
    const float*  __restrict__ bias,
    float*        __restrict__ out32,
    half_t*       __restrict__ out16, int ldo,
    int M, int N, int nvalid, int relu)
{
  const int lane   = threadIdx.x & 31;
  const int wave   = (blockIdx.x * blockDim.x + threadIdx.x) >> 5;
  const int nwaves = (gridDim.x * blockDim.x) >> 5;
  const int ntn    = N >> 4;
  const int ntiles = (M >> 4) * ntn;

  for (int t = wave; t < ntiles; t += nwaves) {
    const int mt = t / ntn, nt = t % ntn;
    v8f acc = {};
    for (int k0 = 0; k0 < k1; k0 += 32) {
      v16h a = load_frag_a(A1, lda1, mt * 16, k0, lane);
      v16h b = load_frag_b(W1, ldw1, nt * 16, k0, lane);
      acc = wmma_f16(a, b, acc);
    }
    if (A2) {
      for (int k0 = 0; k0 < k2; k0 += 32) {
        v16h a = load_frag_a(A2, lda2, mt * 16, k0, lane);
        v16h b = load_frag_b(W2, ldw2, nt * 16, k0, lane);
        acc = wmma_f16(a, b, acc);
      }
    }
    const int   n     = nt * 16 + (lane & 15);
    const float bv    = bias ? bias[n] : 0.0f;
    const int   mbase = mt * 16 + ((lane >> 4) << 3);
#pragma unroll
    for (int r = 0; r < 8; ++r) {
      float v = acc[r] + bv;
      if (relu) v = fmaxf(v, 0.0f);
      if (n < nvalid) {
        const size_t o = (size_t)(mbase + r) * ldo + n;
        if (out32) out32[o] = v;
        if (out16) out16[o] = (half_t)v;
      }
    }
  }
}

// ---------------------------------------------------------------------------
// Attention: scores = scale * (ap[l,b,:] . h[b,:]) masked by length, softmax
// over l, applied[b,:] = sum_l w[l] * enc[l,b,:]. One block per batch row.
// ---------------------------------------------------------------------------
__global__ void attention_kernel(
    const float*  __restrict__ ap,        // [(l*64+b) x 512]
    const float*  __restrict__ hprev,     // [64 x 512]
    const half_t* __restrict__ enc16,     // [(l*64+b) x 512]
    const int*    __restrict__ dlen,      // row 0 of [T x B]
    half_t*       __restrict__ applied16) // [64 x 512]
{
  __shared__ float sc[128];
  __shared__ float red[2];
  const int b = blockIdx.x, tid = threadIdx.x;
  const int lane = tid & 31, wv = tid >> 5;
  const int len = dlen[b];
  const float scale = 0.1f;               // 1/sqrt(L), L=100

  for (int l = wv; l < LL; l += 16) {
    const float* aprow = ap + ((size_t)l * BB + b) * HH;
    const float* hrow  = hprev + (size_t)b * HH;
    float s = 0.0f;
    for (int h = lane; h < HH; h += 32) s += aprow[h] * hrow[h];
#pragma unroll
    for (int m = 16; m; m >>= 1) s += __shfl_xor(s, m, 32);
    if (lane == 0) sc[l] = (l < len) ? s * scale : -INFINITY;
  }
  __syncthreads();
  if (tid < 32) {
    float m = -INFINITY;
    for (int l = lane; l < LL; l += 32) m = fmaxf(m, sc[l]);
#pragma unroll
    for (int k = 16; k; k >>= 1) m = fmaxf(m, __shfl_xor(m, k, 32));
    if (lane == 0) red[0] = m;
  }
  __syncthreads();
  if (tid < LL) sc[tid] = __expf(sc[tid] - red[0]);
  __syncthreads();
  if (tid < 32) {
    float s = 0.0f;
    for (int l = lane; l < LL; l += 32) s += sc[l];
#pragma unroll
    for (int k = 16; k; k >>= 1) s += __shfl_xor(s, k, 32);
    if (lane == 0) red[1] = 1.0f / s;
  }
  __syncthreads();
  {
    const int h = tid;                    // 512 threads <-> H
    float acc = 0.0f;
    for (int l = 0; l < LL; ++l)
      acc += sc[l] * (float)enc16[((size_t)l * BB + b) * HH + h];
    applied16[(size_t)b * HH + h] = (half_t)(acc * red[1]);
  }
}

// ---------------------------------------------------------------------------
// Setup / utility kernels
// ---------------------------------------------------------------------------
__global__ void cvt_f32_to_f16(const float* __restrict__ s, half_t* __restrict__ d, int n) {
  int i = blockIdx.x * blockDim.x + threadIdx.x;
  if (i < n) d[i] = (half_t)s[i];
}
__global__ void fill_f16_zero(half_t* d, int n) {
  int i = blockIdx.x * blockDim.x + threadIdx.x;
  if (i < n) d[i] = (half_t)0.0f;
}
__global__ void fill_f32_zero(float* d, int n) {
  int i = blockIdx.x * blockDim.x + threadIdx.x;
  if (i < n) d[i] = 0.0f;
}
__global__ void copy_f32(const float* __restrict__ s, float* __restrict__ d, int n) {
  int i = blockIdx.x * blockDim.x + threadIdx.x;
  if (i < n) d[i] = s[i];
}
__global__ void add_vec(const float* a, const float* b, float* o, int n) {
  int i = blockIdx.x * blockDim.x + threadIdx.x;
  if (i < n) o[i] = a[i] + b[i];
}
// demonstration[0] is [B,L,D] -> demo16 as [L][B][D] (rows = l*64+b)
__global__ void demo_pack(const float* __restrict__ demo, half_t* __restrict__ out) {
  int i = blockIdx.x * blockDim.x + threadIdx.x;
  if (i < LL * BB * DD) {
    const int d = i & (DD - 1);
    const int b = (i >> 7) & 63;
    const int l = i >> 13;
    out[i] = (half_t)demo[((size_t)b * LL + l) * DD + d];
  }
}
// Fused head: Wq[a,k] = sum_j outW[a,j]*midW[j,k] ; bq[a]=outb[a]+outW[a,:].midb
__global__ void build_wq(const float* __restrict__ midW, const float* __restrict__ midb,
                         const float* __restrict__ outW, const float* __restrict__ outb,
                         half_t* __restrict__ wq, float* __restrict__ bq) {
  const int a = blockIdx.x;   // 0..31 (rows 18..31 zero-padded)
  const int h = threadIdx.x;  // 0..511
  if (a >= AA) { wq[(size_t)a * HH + h] = (half_t)0.0f; if (h == 0) bq[a] = 0.0f; return; }
  float s = 0.0f;
  for (int j = 0; j < HH; ++j) s += outW[(size_t)a * HH + j] * midW[(size_t)j * HH + h];
  wq[(size_t)a * HH + h] = (half_t)s;
  if (h == 0) {
    float t = outb[a];
    for (int j = 0; j < HH; ++j) t += outW[(size_t)a * HH + j] * midb[j];
    bq[a] = t;
  }
}

// ---------------------------------------------------------------------------
// Host orchestration
// ---------------------------------------------------------------------------
extern "C" void kernel_launch(void* const* d_in, const int* in_sizes, int n_in,
                              void* d_out, int out_size, void* d_ws, size_t ws_size,
                              hipStream_t stream) {
  (void)in_sizes; (void)n_in; (void)out_size; (void)ws_size;

  const float* state   = (const float*)d_in[0];
  const float* demo    = (const float*)d_in[1];
  const int*   dlen    = (const int*)d_in[2];
  const float* h0      = (const float*)d_in[5];
  const float* c0      = (const float*)d_in[6];
  const float* encWih  = (const float*)d_in[7];
  const float* encWhh  = (const float*)d_in[8];
  const float* encbih  = (const float*)d_in[9];
  const float* encbhh  = (const float*)d_in[10];
  const float* attnW   = (const float*)d_in[11];
  const float* attnb   = (const float*)d_in[12];
  const float* combW   = (const float*)d_in[13];
  const float* combb   = (const float*)d_in[14];
  const float* lstmWih = (const float*)d_in[15];
  const float* lstmWhh = (const float*)d_in[16];
  const float* lstmbih = (const float*)d_in[17];
  const float* lstmbhh = (const float*)d_in[18];
  const float* midW    = (const float*)d_in[19];
  const float* midb    = (const float*)d_in[20];
  const float* outW    = (const float*)d_in[21];
  const float* outb    = (const float*)d_in[22];

  float* q    = (float*)d_out;                 // [T,B,A]
  float* hout = q + (size_t)TT * BB * AA;      // [B,H]
  float* cout = hout + (size_t)BB * HH;        // [B,H]

  // ---- workspace carve-up -------------------------------------------------
  char* ws = (char*)d_ws;
  size_t off = 0;
  auto alloc = [&](size_t bytes) -> void* {
    void* p = ws + off;
    off = (off + bytes + 255) & ~(size_t)255;
    return p;
  };
  half_t* demo16   = (half_t*)alloc((size_t)LL * BB * DD * 2);
  half_t* state16  = (half_t*)alloc((size_t)TT * BB * DD * 2);
  half_t* encWih16 = (half_t*)alloc((size_t)G4H * DD * 2);
  half_t* encWhh16 = (half_t*)alloc((size_t)G4H * HH * 2);
  half_t* decWih16 = (half_t*)alloc((size_t)G4H * HH * 2);
  half_t* decWhh16 = (half_t*)alloc((size_t)G4H * HH * 2);
  half_t* attnW16  = (half_t*)alloc((size_t)HH * HH * 2);
  half_t* combW16  = (half_t*)alloc((size_t)HH * (DD + HH) * 2);
  half_t* wq16     = (half_t*)alloc((size_t)32 * HH * 2);
  float*  benc     = (float*)alloc((size_t)G4H * 4);
  float*  bdec     = (float*)alloc((size_t)G4H * 4);
  float*  bq       = (float*)alloc((size_t)32 * 4);
  half_t* enc16    = (half_t*)alloc((size_t)LL * BB * HH * 2);
  half_t* zeros16  = (half_t*)alloc((size_t)BB * HH * 2);
  float*  cenc     = (float*)alloc((size_t)BB * HH * 4);
  float*  ap       = (float*)alloc((size_t)LL * BB * HH * 4);
  half_t* hdec16a  = (half_t*)alloc((size_t)BB * HH * 2);
  half_t* hdec16b  = (half_t*)alloc((size_t)BB * HH * 2);
  float*  hdec32   = (float*)alloc((size_t)BB * HH * 4);
  float*  cdec     = (float*)alloc((size_t)BB * HH * 4);
  half_t* applied16= (half_t*)alloc((size_t)BB * HH * 2);
  half_t* xcomb16  = (half_t*)alloc((size_t)BB * HH * 2);

#define LAUNCH1D(K, n, ...) K<<<((n) + 255) / 256, 256, 0, stream>>>(__VA_ARGS__)

  // ---- setup: f16 conversions, bias sums, fused head, init state ----------
  LAUNCH1D(cvt_f32_to_f16, G4H * DD, encWih,  encWih16, G4H * DD);
  LAUNCH1D(cvt_f32_to_f16, G4H * HH, encWhh,  encWhh16, G4H * HH);
  LAUNCH1D(cvt_f32_to_f16, G4H * HH, lstmWih, decWih16, G4H * HH);
  LAUNCH1D(cvt_f32_to_f16, G4H * HH, lstmWhh, decWhh16, G4H * HH);
  LAUNCH1D(cvt_f32_to_f16, HH * HH,  attnW,   attnW16,  HH * HH);
  LAUNCH1D(cvt_f32_to_f16, HH * (DD + HH), combW, combW16, HH * (DD + HH));
  LAUNCH1D(cvt_f32_to_f16, TT * BB * DD, state, state16, TT * BB * DD);
  LAUNCH1D(cvt_f32_to_f16, BB * HH, h0, hdec16a, BB * HH);
  LAUNCH1D(demo_pack, LL * BB * DD, demo, demo16);
  LAUNCH1D(add_vec, G4H, encbih, encbhh, benc, G4H);
  LAUNCH1D(add_vec, G4H, lstmbih, lstmbhh, bdec, G4H);
  build_wq<<<32, 512, 0, stream>>>(midW, midb, outW, outb, wq16, bq);
  LAUNCH1D(fill_f16_zero, BB * HH, zeros16, BB * HH);
  LAUNCH1D(fill_f32_zero, BB * HH, cenc, BB * HH);
  LAUNCH1D(copy_f32, BB * HH, h0, hdec32, BB * HH);
  LAUNCH1D(copy_f32, BB * HH, c0, cdec, BB * HH);

  // ---- encoder scan over L (sequential; 512 WMMA tiles / step) ------------
  for (int l = 0; l < LL; ++l) {
    const half_t* a2 = (l == 0) ? zeros16 : enc16 + (size_t)(l - 1) * BB * HH;
    lstm_step_kernel<<<32, 512, 0, stream>>>(
        demo16 + (size_t)l * BB * DD, DD, DD, encWih16,
        a2, encWhh16, benc, cenc,
        enc16 + (size_t)l * BB * HH, nullptr);
  }

  // ---- attn_proj = enc @ attn_W^T + attn_b  (6400 x 512 x 512) ------------
  gemm_wmma_kernel<<<800, 512, 0, stream>>>(
      enc16, HH, HH, attnW16, HH,
      nullptr, 0, 0, nullptr, 0,
      attnb, ap, nullptr, HH, LL * BB, HH, HH, 0);

  // ---- decoder scan over T ------------------------------------------------
  for (int t = 0; t < TT; ++t) {
    const half_t* hin16  = (t & 1) ? hdec16b : hdec16a;
    half_t*       hout16 = (t & 1) ? hdec16a : hdec16b;

    attention_kernel<<<BB, 512, 0, stream>>>(ap, hdec32, enc16, dlen, applied16);

    // x = relu([applied | state_t] @ comb_W^T + comb_b)
    gemm_wmma_kernel<<<8, 512, 0, stream>>>(
        applied16, HH, HH, combW16, DD + HH,
        state16 + (size_t)t * BB * DD, DD, DD, combW16 + HH, DD + HH,
        combb, nullptr, xcomb16, HH, BB, HH, HH, 1);

    lstm_step_kernel<<<32, 512, 0, stream>>>(
        xcomb16, HH, HH, decWih16,
        hin16, decWhh16, bdec, cdec, hout16, hdec32);

    // q_t = h @ (out_W mid_W)^T + fused bias  (N padded 18->32)
    gemm_wmma_kernel<<<1, 512, 0, stream>>>(
        hout16, HH, HH, wq16, HH,
        nullptr, 0, 0, nullptr, 0,
        bq, q + (size_t)t * BB * AA, nullptr, AA, BB, 32, AA, 0);
  }

  // ---- final h, c ---------------------------------------------------------
  LAUNCH1D(copy_f32, BB * HH, hdec32, hout, BB * HH);
  LAUNCH1D(copy_f32, BB * HH, cdec, cout, BB * HH);
#undef LAUNCH1D
}
